// SelectiveScan_47536698032518
// MI455X (gfx1250) — matrix operand order repaired and checked by reading, not compiled, not run on hardware
//
#include <hip/hip_runtime.h>

// ---------------- CDNA5 WMMA fragment types (wave32) ----------------
typedef __attribute__((ext_vector_type(16))) _Float16 v16h;
typedef __attribute__((ext_vector_type(8)))  float    v8f;
typedef __attribute__((ext_vector_type(4)))  uint32_t u32x4;
typedef __attribute__((ext_vector_type(8)))  uint32_t u32x8;

#define LN_EPS 1e-5f

// Problem dims (fixed by the reference)
#define B_   4
#define C_   192
#define T_   32
#define H_   32
#define W_   32
#define DS   32
#define HW   1024            // H_*W_

#define SITES   16           // site tile per block (16 consecutive ww)
#define THREADS 128          // 4 waves
#define XLD     200          // padded leading dim for 192-wide output LDS tile

// fast sigmoid: v_exp_f32 + raw v_rcp_f32 (avoids IEEE div expansion)
__device__ inline float fast_sigmoid(float x) {
    return __builtin_amdgcn_rcpf(1.0f + __expf(-x));
}

struct Smem {
    // f16 normalized input, pre-swizzled into WMMA A-fragment order:
    // xh[kk][g][row][u*8+v]  <->  K = kk*32 + u*16 + g*8 + v ; one v16h = 32B contiguous
    __attribute__((aligned(32))) _Float16 xh[6][2][SITES][16];
    float xraw[2][C_][SITES]; // TDM destination, double buffered ([ch][site], dim0 contiguous)
    float proj[SITES][64];    // in_proj output (x_proj | gate_logits)
    float states[SITES][DS];  // recurrence states (shared between waves)
    float comb[SITES][DS];    // states@B^T + x_proj@C^T
    float outt[SITES][XLD];   // output tile (16 x 192)
    float red_s[8][SITES];    // LN partial sums
    float red_q[8][SITES];    // LN partial sumsq
    float mu[SITES];
    float rs[SITES];
    float gam[C_], bet[C_], obz[C_];
};

// ---- Tensor Data Mover: 2D tile (16 floats x 192 rows, row stride 32768 elts) -> LDS ----
__device__ inline void tdm_load_tile(uint32_t lds_addr, const float* gptr) {
    uint64_t ga = (uint64_t)(uintptr_t)gptr;
    u32x4 g0;
    g0[0] = 1u;                                   // [1:0] count=1 (valid user D#)
    g0[1] = lds_addr;                             // [63:32] LDS byte address
    g0[2] = (uint32_t)ga;                         // [95:64] global_addr lo
    g0[3] = (uint32_t)((ga >> 32) & 0x1FFFFFFu)   // [120:96] global_addr hi
          | (2u << 30);                           // [127:126] type = 2 (image)
    u32x8 g1;
    g1[0] = (2u << 16);                           // data_size=2 (4 bytes); wg_mask=0
    g1[1] = (16u << 16);                          // tensor_dim0[15:0]=16 ; atomic_barrier=0
    g1[2] = ((uint32_t)C_ << 16);                 // tensor_dim1[15:0]=192
    g1[3] = (16u << 16);                          // tile_dim0=16
    g1[4] = (uint32_t)C_;                         // tile_dim1=192, tile_dim2=0
    g1[5] = (uint32_t)(T_ * HW);                  // tensor_dim0_stride = 32768 elements
    g1[6] = 0u;                                   // stride hi, tensor_dim1_stride lo
    g1[7] = 0u;
    u32x4 g2 = {0u, 0u, 0u, 0u};                  // dims 2/3 unused (tile_dim3=0)
    u32x4 g3 = {0u, 0u, 0u, 0u};
    asm volatile("tensor_load_to_lds %0, %1, %2, %3"
                 :: "s"(g0), "s"(g1), "s"(g2), "s"(g3)
                 : "memory");
}

// A-fragment (16x32, f16) from fp32 LDS: lane = 16*g + row; e -> K = 16*(e/8)+8*g+(e&7)
__device__ inline v16h ld_a_frag(const float* base, int ld, int lane) {
    int row = lane & 15, g = lane >> 4;
    const float* p = base + row * ld;
    v16h a;
#pragma unroll
    for (int e = 0; e < 16; ++e) {
        int k = ((e >> 3) << 4) + (g << 3) + (e & 7);
        a[e] = (_Float16)p[k];
    }
    return a;
}

// B-fragment (32x16, f16) from a row-major weight W where B[k][n] = W[ncol0+n][k].
__device__ inline v16h ld_b_frag(const float* __restrict__ w, int ldw, int ncol0, int lane) {
    int n = lane & 15, g = lane >> 4;
    const float* p = w + (size_t)(ncol0 + n) * ldw;
    v16h b;
#pragma unroll
    for (int e = 0; e < 16; ++e) b[e] = (_Float16)p[((e >> 3) << 4) + (g << 3) + (e & 7)];
    return b;
}

__global__ __launch_bounds__(THREADS)
void selscan_fused_kernel(const float* __restrict__ x,
                          const float* __restrict__ in_w,  const float* __restrict__ in_b,
                          const float* __restrict__ out_w, const float* __restrict__ out_b,
                          const float* __restrict__ Am,    const float* __restrict__ Bm,
                          const float* __restrict__ Cm,    const float* __restrict__ Dm,
                          const float* __restrict__ gamma, const float* __restrict__ beta,
                          float* __restrict__ out)
{
    __shared__ Smem sm;

    const int tid  = threadIdx.x;
    const int lane = tid & 31;
    const int wv   = tid >> 5;        // 0..3
    const int wcol = wv & 1;          // waves {0,2},{1,3} duplicate the ds-column tiles
    const int n_l  = lane & 15;
    const int g_l  = lane >> 4;

    const int blk  = blockIdx.x;      // 256 blocks
    const int ww0  = (blk & 1) * 16;
    const int hh   = (blk >> 1) & 31;
    const int bb   = blk >> 6;
    const int plane = hh * W_ + ww0;

    // ---- stage small per-channel constants into LDS ----
    for (int i = tid; i < C_; i += THREADS) {
        sm.gam[i] = gamma[i];
        sm.bet[i] = beta[i];
        sm.obz[i] = out_b[i];
    }

    // ---- preload all weight B-fragments into registers (reused for all t) ----
    v16h inw_f[6];
#pragma unroll
    for (int kk = 0; kk < 6; ++kk)
        inw_f[kk] = ld_b_frag(in_w + kk * 32, C_, wv * 16, lane);

    v16h outw_f[3];
#pragma unroll
    for (int s = 0; s < 3; ++s)
        outw_f[s] = ld_b_frag(out_w, DS, (wv * 3 + s) * 16, lane);

    v16h Af = ld_b_frag(Am, DS, wcol * 16, lane);
    v16h Bf = ld_b_frag(Bm, DS, wcol * 16, lane);
    v16h Cf = ld_b_frag(Cm, DS, wcol * 16, lane);

    const float inb_l   = in_b[wv * 16 + n_l];
    const float decay_l = fast_sigmoid(Dm[wcol * 16 + n_l]);

    v8f state = {};   // persistent recurrence state fragment

    const int site = tid & 15;
    const int part = tid >> 4;     // 0..7

    const float* tile0 = x + ((size_t)bb * C_ * T_) * HW + plane;   // + t*HW per step

    // kick off TDM for t = 0 (wave 0 issues; TDM ignores EXEC, counts on TENSORcnt)
    if (wv == 0)
        tdm_load_tile((uint32_t)(uintptr_t)(void*)&sm.xraw[0][0][0], tile0);

    __syncthreads();

    int buf = 0;
    for (int t = 0; t < T_; ++t) {
        // ---------- wait for TDM tile t, then immediately launch tile t+1 ----------
        if (wv == 0) {
            __builtin_amdgcn_s_wait_tensorcnt(0);
        }
        __syncthreads();    // TDM data in sm.xraw[buf] now visible to all waves
        if (wv == 0 && (t + 1) < T_)
            tdm_load_tile((uint32_t)(uintptr_t)(void*)&sm.xraw[buf ^ 1][0][0],
                          tile0 + (size_t)(t + 1) * HW);

        // ---------- LN statistics (values cached in registers) ----------
        const float* raw = &sm.xraw[buf][0][0];
        float vals[24];
        float s0 = 0.f, s1 = 0.f;
#pragma unroll
        for (int i = 0; i < 12; ++i) {
            int ch = (part + (i << 3)) * 2;
            float v0 = raw[ch * SITES + site];
            float v1 = raw[(ch + 1) * SITES + site];
            vals[2 * i]     = v0;
            vals[2 * i + 1] = v1;
            s0 += v0 + v1;
            s1 += v0 * v0 + v1 * v1;
        }
        sm.red_s[part][site] = s0;
        sm.red_q[part][site] = s1;
        __syncthreads();
        if (tid < SITES) {
            float a = 0.f, q = 0.f;
#pragma unroll
            for (int p = 0; p < 8; ++p) { a += sm.red_s[p][tid]; q += sm.red_q[p][tid]; }
            float mu  = a * (1.0f / C_);
            float var = q * (1.0f / C_) - mu * mu;
            sm.mu[tid] = mu;
            sm.rs[tid] = rsqrtf(var + LN_EPS);
        }
        __syncthreads();

        // ---------- normalize + convert f16 ONCE, store in WMMA A-fragment order ----------
        {
            float mu = sm.mu[site], rs = sm.rs[site];
#pragma unroll
            for (int i = 0; i < 12; ++i) {
#pragma unroll
                for (int d = 0; d < 2; ++d) {
                    int ch = (part + (i << 3)) * 2 + d;
                    float v = (vals[2 * i + d] - mu) * rs * sm.gam[ch] + sm.bet[ch];
                    int kk = ch >> 5, r = ch & 31;
                    int u = r >> 4, g = (r >> 3) & 1, vv = r & 7;
                    sm.xh[kk][g][site][u * 8 + vv] = (_Float16)v;
                }
            }
        }
        __syncthreads();

        // ---------- stage 1: in_proj; A-fragment = one contiguous 32B LDS read ----------
        {
            v8f acc = {};
#pragma unroll
            for (int kk = 0; kk < 6; ++kk) {
                v16h a = *(const v16h*)&sm.xh[kk][g_l][n_l][0];
                acc = __builtin_amdgcn_wmma_f32_16x16x32_f16(
                        false, a, false, inw_f[kk], (short)0, acc, false, false);
            }
#pragma unroll
            for (int j = 0; j < 8; ++j)
                sm.proj[j + g_l * 8][wv * 16 + n_l] = acc[j] + inb_l;
        }
        __syncthreads();

        // ---------- stage 2: u = (x_proj @ A^T) * sigmoid(gate); state update ----------
        v16h a_xp = ld_a_frag(&sm.proj[0][0], 64, lane);
        {
            v8f z = {};
            v8f u = __builtin_amdgcn_wmma_f32_16x16x32_f16(
                        false, a_xp, false, Af, (short)0, z, false, false);
#pragma unroll
            for (int j = 0; j < 8; ++j) {
                int m = j + g_l * 8;
                float gate = fast_sigmoid(sm.proj[m][32 + wcol * 16 + n_l]);
                state[j] = state[j] * decay_l + u[j] * gate;
                sm.states[m][wcol * 16 + n_l] = state[j];
            }
        }
        __syncthreads();

        // ---------- stage 3: comb = states @ B^T + x_proj @ C^T ----------
        {
            v16h a_st = ld_a_frag(&sm.states[0][0], DS, lane);
            v8f z = {};
            v8f acc = __builtin_amdgcn_wmma_f32_16x16x32_f16(
                        false, a_st, false, Bf, (short)0, z, false, false);
            acc = __builtin_amdgcn_wmma_f32_16x16x32_f16(
                        false, a_xp, false, Cf, (short)0, acc, false, false);
#pragma unroll
            for (int j = 0; j < 8; ++j)
                sm.comb[j + g_l * 8][wcol * 16 + n_l] = acc[j];
        }
        __syncthreads();

        // ---------- stage 4: out = comb @ out_w^T; 3 column tiles per wave ----------
        {
            v16h a_cb = ld_a_frag(&sm.comb[0][0], DS, lane);
#pragma unroll
            for (int s = 0; s < 3; ++s) {
                v8f z = {};
                v8f acc = __builtin_amdgcn_wmma_f32_16x16x32_f16(
                            false, a_cb, false, outw_f[s], (short)0, z, false, false);
                int ct = wv * 3 + s;
#pragma unroll
                for (int j = 0; j < 8; ++j)
                    sm.outt[j + g_l * 8][ct * 16 + n_l] = acc[j];
            }
        }
        __syncthreads();

        // ---------- coalesced store (lanes 0..15 hit 16 consecutive floats) ----------
        const size_t tbase = ((size_t)bb * C_ * T_ + t) * HW + plane + site;
#pragma unroll 1
        for (int i = 0; i < 24; ++i) {
            int ch = part + (i << 3);
            out[tbase + (size_t)ch * (T_ * HW)] = sm.outt[site][ch] + sm.obz[ch];
        }
        __syncthreads();

        buf ^= 1;
    }
}

extern "C" void kernel_launch(void* const* d_in, const int* in_sizes, int n_in,
                              void* d_out, int out_size, void* d_ws, size_t ws_size,
                              hipStream_t stream) {
    (void)in_sizes; (void)n_in; (void)out_size; (void)d_ws; (void)ws_size;
    const float* x     = (const float*)d_in[0];
    const float* in_w  = (const float*)d_in[1];
    const float* in_b  = (const float*)d_in[2];
    const float* out_w = (const float*)d_in[3];
    const float* out_b = (const float*)d_in[4];
    const float* Am    = (const float*)d_in[5];
    const float* Bm    = (const float*)d_in[6];
    const float* Cm    = (const float*)d_in[7];
    const float* Dm    = (const float*)d_in[8];
    const float* gam   = (const float*)d_in[9];
    const float* bet   = (const float*)d_in[10];
    float* out = (float*)d_out;

    dim3 grid(256), block(THREADS);
    selscan_fused_kernel<<<grid, block, 0, stream>>>(
        x, in_w, in_b, out_w, out_b, Am, Bm, Cm, Dm, gam, bet, out);
}
